// GCNEncoder_35605278883994
// MI455X (gfx1250) — compile-verified, matrix operand order checked by reading, not run on hardware
//
#include <hip/hip_runtime.h>

// ---------------------------------------------------------------------------
// GCN 2-layer encoder for MI455X (gfx1250, wave32, WMMA).
//  GEMMs: bf16 WMMA (v_wmma_f32_16x16x32_bf16, fp32 accum); each wave keeps
//  its full 16-row x K=128 A strip in registers (32 VGPRs) and sweeps all
//  column tiles -> A loaded exactly once, B (<=32KB) L0-resident.
//  Edge aggregation: one wave per edge, 128-bit gathers + f32 atomic scatter
//  (h and agg buffers are L2-resident: 192MB L2 >> 51MB working set).
// ---------------------------------------------------------------------------

typedef __attribute__((ext_vector_type(16))) __bf16 v16bf;
typedef __attribute__((ext_vector_type(8)))  __bf16 v8bf;
typedef __attribute__((ext_vector_type(8)))  float  v8f;

// ------------------------- prep / elementwise ------------------------------

__global__ void k_init_deg(float* __restrict__ deg, int n) {
  int i = blockIdx.x * blockDim.x + threadIdx.x;
  if (i < n) deg[i] = 1.0f;  // self-loop contributes 1 to in-degree
}

__global__ void k_deg_accum(const long long* __restrict__ dst,
                            float* __restrict__ deg, int nE) {
  int e = blockIdx.x * blockDim.x + threadIdx.x;
  if (e < nE) atomicAdd(&deg[(int)dst[e]], 1.0f);
}

__global__ void k_dinv(const float* __restrict__ deg,
                       float* __restrict__ dinv, int n) {
  int i = blockIdx.x * blockDim.x + threadIdx.x;
  if (i < n) dinv[i] = rsqrtf(deg[i]);  // deg >= 1 always (self-loop)
}

__global__ void k_cvt_bf16(const float* __restrict__ x,
                           __bf16* __restrict__ y, int n) {
  int i = blockIdx.x * blockDim.x + threadIdx.x;
  if (i < n) y[i] = (__bf16)x[i];
}

// W is [IN][OUT] row-major; produce Wt[OUT][IN] bf16 so the WMMA B operand
// (columns of W) is contiguous along K.
__global__ void k_transpose_cvt(const float* __restrict__ W,
                                __bf16* __restrict__ Wt, int IN, int OUT) {
  int i = blockIdx.x * blockDim.x + threadIdx.x;
  if (i >= IN * OUT) return;
  int no = i / IN, k = i % IN;
  Wt[i] = (__bf16)W[k * OUT + no];
}

// agg[i, :] = dinv[i]^2 * h[i, :]   (self-loop message; full overwrite)
__global__ void k_self_term(const float* __restrict__ h,
                            const float* __restrict__ dinv,
                            float* __restrict__ agg, int n, int C) {
  int i = blockIdx.x * blockDim.x + threadIdx.x;
  if (i >= n * C) return;
  float w = dinv[i / C];
  agg[i] = w * w * h[i];
}

__global__ void k_bias_relu_cvt(const float* __restrict__ agg,
                                const float* __restrict__ b,
                                __bf16* __restrict__ out, int n, int C) {
  int i = blockIdx.x * blockDim.x + threadIdx.x;
  if (i >= n * C) return;
  float v = agg[i] + b[i % C];
  out[i] = (__bf16)(v > 0.0f ? v : 0.0f);
}

__global__ void k_bias_add(float* __restrict__ out,
                           const float* __restrict__ b, int n, int C) {
  int i = blockIdx.x * blockDim.x + threadIdx.x;
  if (i >= n * C) return;
  out[i] += b[i % C];
}

// ------------------------- WMMA GEMM: C = A(bf16) x Bt(bf16)^T -------------
// A: [nrows x IN] row-major bf16.  Bt: [OUT x IN] row-major bf16 (i.e. W^T).
// One wave owns a 16-row strip: loads the full K=IN A strip into registers
// once, then sweeps all OUT/16 column tiles (A-reuse factor OUT/16).
//
// Operand layout (CDNA5 ISA 7.12.2, 16-bit A 16x32):
//   lane m (0..15):  elems 0..7 -> K = k0+0..7,   elems 8..15 -> K = k0+16..23
//   lane m+16:       elems 0..7 -> K = k0+8..15,  elems 8..15 -> K = k0+24..31
// B operand = A-layout applied to B^T rows (columns of W), contiguous in K.
template <int IN, int OUT>
__global__ void gemm_wmma_bf16(const __bf16* __restrict__ A,
                               const __bf16* __restrict__ Bt,
                               float* __restrict__ C, int nrows) {
  constexpr int KT = IN / 32;  // k-steps of 32
  int wave = (int)((blockIdx.x * blockDim.x + threadIdx.x) >> 5);
  int lane = threadIdx.x & 31;
  int row0 = wave * 16;
  if (row0 >= nrows) return;  // nrows must be a multiple of 16 (50000 is)

  int half = lane >> 4;   // 0: lanes 0-15, 1: lanes 16-31
  int m    = lane & 15;

  // ---- load entire A strip for this wave (KT * 8 VGPRs) ----
  const __bf16* arow = A + (size_t)(row0 + m) * IN;
  v16bf a[KT];
#pragma unroll
  for (int j = 0; j < KT; ++j) {
    v8bf lo = *(const v8bf*)(arow + 32 * j + half * 8);
    v8bf hi = *(const v8bf*)(arow + 32 * j + 16 + half * 8);
#pragma unroll
    for (int i = 0; i < 8; ++i) { a[j][i] = lo[i]; a[j][8 + i] = hi[i]; }
  }

  // ---- sweep all column tiles against the register-resident A strip ----
#pragma unroll
  for (int ct = 0; ct < OUT / 16; ++ct) {
    const __bf16* brow = Bt + (size_t)(ct * 16 + m) * IN;
    v8f acc = {};
#pragma unroll
    for (int j = 0; j < KT; ++j) {
      v8bf lo = *(const v8bf*)(brow + 32 * j + half * 8);
      v8bf hi = *(const v8bf*)(brow + 32 * j + 16 + half * 8);
      v16bf b;
#pragma unroll
      for (int i = 0; i < 8; ++i) { b[i] = lo[i]; b[8 + i] = hi[i]; }
      acc = __builtin_amdgcn_wmma_f32_16x16x32_bf16(
          /*neg_a=*/false, a[j], /*neg_b=*/false, b,
          /*c_mod=*/(short)0, acc, /*reuse_a=*/false, /*reuse_b=*/false);
    }
    // C/D layout: VGPR r, lanes 0-15 -> M=r, N=lane; lanes 16-31 -> M=8+r
#pragma unroll
    for (int r = 0; r < 8; ++r)
      C[(size_t)(row0 + half * 8 + r) * OUT + ct * 16 + m] = acc[r];
  }
}

// ------------------------- edge gather/scatter -----------------------------
// One wave per edge; each lane owns C/32 consecutive channels (128-bit load
// for C=128). Scatter-add via global_atomic_add_f32 (resolved in L2).
template <int C>
__global__ void edge_scatter(const long long* __restrict__ src,
                             const long long* __restrict__ dst,
                             const float* __restrict__ dinv,
                             const float* __restrict__ H,
                             float* __restrict__ Agg, int nE) {
  constexpr int V = C / 32;
  int wid  = (int)((blockIdx.x * blockDim.x + threadIdx.x) >> 5);
  int lane = threadIdx.x & 31;
  if (wid >= nE) return;
  int s = (int)src[wid];
  int d = (int)dst[wid];
  float w = dinv[s] * dinv[d];
  const float* hp = H   + (size_t)s * C + lane * V;
  float*       ap = Agg + (size_t)d * C + lane * V;
#pragma unroll
  for (int v = 0; v < V; ++v) atomicAdd(ap + v, w * hp[v]);
}

// ---------------------------------------------------------------------------

extern "C" void kernel_launch(void* const* d_in, const int* in_sizes, int n_in,
                              void* d_out, int out_size, void* d_ws, size_t ws_size,
                              hipStream_t stream) {
  (void)n_in; (void)out_size; (void)ws_size;

  const float*     x   = (const float*)d_in[0];
  const long long* ei  = (const long long*)d_in[1];  // int64 per reference
  const float*     W1  = (const float*)d_in[2];
  const float*     b1  = (const float*)d_in[3];
  const float*     W2  = (const float*)d_in[4];
  const float*     b2  = (const float*)d_in[5];
  float*           out = (float*)d_out;

  const int IN  = 128, HID = 128, Z = 64;
  const int n  = in_sizes[0] / IN;     // 50000
  const int nE = in_sizes[1] / 2;      // 800000
  const long long* srcp = ei;          // edge_index[0]
  const long long* dstp = ei + nE;     // edge_index[1]

  // ---- workspace carve-out (256B aligned) ----
  char* ws = (char*)d_ws;
  size_t off = 0;
  auto take = [&](size_t bytes) -> char* {
    char* p = ws + off;
    off = (off + bytes + 255) & ~(size_t)255;
    return p;
  };
  float*  deg  = (float*)take((size_t)n * 4);
  float*  dinv = (float*)take((size_t)n * 4);
  __bf16* w1t  = (__bf16*)take((size_t)IN * HID * 2);
  __bf16* w2t  = (__bf16*)take((size_t)HID * Z * 2);
  __bf16* xb   = (__bf16*)take((size_t)n * IN * 2);   // layer-1 A; reused as hb
  float*  h    = (float*)take((size_t)n * HID * 4);   // h1; reused as h2 (n*Z)
  float*  agg  = (float*)take((size_t)n * HID * 4);   // layer-1 aggregate

  auto blk = [](long long t) { return (unsigned)((t + 255) / 256); };

  // ---- degree / norm ----
  k_init_deg<<<blk(n), 256, 0, stream>>>(deg, n);
  k_deg_accum<<<blk(nE), 256, 0, stream>>>(dstp, deg, nE);
  k_dinv<<<blk(n), 256, 0, stream>>>(deg, dinv, n);

  // ---- operand prep (fp32 -> bf16, W transposed for contiguous-K B) ----
  k_cvt_bf16<<<blk((long long)n * IN), 256, 0, stream>>>(x, xb, n * IN);
  k_transpose_cvt<<<blk(IN * HID), 256, 0, stream>>>(W1, w1t, IN, HID);
  k_transpose_cvt<<<blk(HID * Z), 256, 0, stream>>>(W2, w2t, HID, Z);

  const long long waves = (n + 15) / 16;  // one wave per 16-row strip

  // ---- layer 1: h = x @ W1 (WMMA), then norm-scatter, bias+relu ----
  gemm_wmma_bf16<128, 128><<<blk(waves * 32), 256, 0, stream>>>(xb, w1t, h, n);
  k_self_term<<<blk((long long)n * HID), 256, 0, stream>>>(h, dinv, agg, n, HID);
  edge_scatter<128><<<blk((long long)nE * 32), 256, 0, stream>>>(srcp, dstp, dinv, h, agg, nE);
  k_bias_relu_cvt<<<blk((long long)n * HID), 256, 0, stream>>>(agg, b1, xb, n, HID);

  // ---- layer 2: h2 = h1 @ W2 (WMMA), norm-scatter into d_out, +b2 ----
  gemm_wmma_bf16<128, 64><<<blk(waves * 32), 256, 0, stream>>>(xb, w2t, h, n);
  k_self_term<<<blk((long long)n * Z), 256, 0, stream>>>(h, dinv, out, n, Z);
  edge_scatter<64><<<blk((long long)nE * 32), 256, 0, stream>>>(srcp, dstp, dinv, h, out, nE);
  k_bias_add<<<blk((long long)n * Z), 256, 0, stream>>>(out, b2, n, Z);
}
